// NeuralProcessConv_24343874634007
// MI455X (gfx1250) — matrix operand order repaired and checked by reading, not compiled, not run on hardware
//
#include <hip/hip_runtime.h>
#include <math.h>

#define GRIDW 50
#define YD    2500
#define BB    16
#define NCV   64
#define NTGT  128
#define CC    32
#define RD    128
#define ZD    64
#define HD    128
#define NTILES 157   // ceil(2500/16)

typedef __attribute__((ext_vector_type(16))) __bf16        v16bf;
typedef __attribute__((ext_vector_type(8)))  float         v8f;
typedef __attribute__((ext_vector_type(8)))  unsigned int  v8u;

// Build a 16xbf16 WMMA fragment from two 16-byte register quads (no memory).
__device__ __forceinline__ v16bf mk_frag(uint4 a, uint4 b) {
  v8u u = {a.x, a.y, a.z, a.w, b.x, b.y, b.z, b.w};
  return __builtin_bit_cast(v16bf, u);
}

__device__ __forceinline__ unsigned short f2bf(float f) {
  union { float f; unsigned int u; } c; c.f = f;
  unsigned int r = c.u + 0x7FFFu + ((c.u >> 16) & 1u);
  return (unsigned short)(r >> 16);
}

// ---------------- workspace byte offsets ----------------
// racc : 16*128 f32          @ 0        (8192 B)
// z    : 16*64  f32          @ 8192     (4096 B)
// h2   : 2048*128 bf16       @ 12288    (524288 B)
// wf2  : 9*2*32*16 bf16      @ 536576   (18432 B)
// fmu  : 157*4*32*16 bf16    @ 555008   (643072 B)
// fsig : 157*4*32*16 bf16    @ 1198080  (643072 B)

__global__ void k_init(float* racc) {
  int i = blockIdx.x * blockDim.x + threadIdx.x;
  if (i < BB * RD) racc[i] = 0.f;
}

// Relayout conv2 weights into WMMA B-fragment order.
__global__ void k_prep_w2(const float* __restrict__ w2, unsigned short* __restrict__ wf) {
  int idx = blockIdx.x * blockDim.x + threadIdx.x;
  if (idx >= 9 * 2 * 32 * 16) return;
  int j    =  idx        & 15;
  int lane = (idx >> 4)  & 31;
  int nt   = (idx >> 9)  & 1;
  int tap  =  idx >> 10;
  int cout = nt * 16 + (lane & 15);
  int cin  = (lane >> 4) * 16 + j;
  wf[idx] = f2bf(w2[(cout * CC + cin) * 9 + tap]);
}

// Relayout Wdmu/Wdsig [128,2500] into B-fragment order (zero-pad cols >= 2500).
__global__ void k_prep_wout(const float* __restrict__ Wmu, const float* __restrict__ Wsig,
                            unsigned short* __restrict__ fmu, unsigned short* __restrict__ fsig) {
  const int PER = NTILES * 4 * 32 * 16;  // 321536
  int idx = blockIdx.x * blockDim.x + threadIdx.x;
  if (idx >= 2 * PER) return;
  const float* W = (idx < PER) ? Wmu : Wsig;
  unsigned short* F = (idx < PER) ? fmu : fsig;
  int r = (idx < PER) ? idx : idx - PER;
  int j     =  r        & 15;
  int lane  = (r >> 4)  & 31;
  int kb    = (r >> 9)  & 3;
  int ntile =  r >> 11;
  int col = ntile * 16 + (lane & 15);
  int k   = kb * 32 + (lane >> 4) * 16 + j;
  float v = (col < YD) ? W[k * YD + col] : 0.f;
  F[r] = f2bf(v);
}

// One workgroup per context sample (1024 blocks).
// conv1 (VALU) -> LDS bf16 [52][52][32] (zero halo); conv2 = implicit GEMM on
// v_wmma_f32_16x16x32_bf16 fused with the mean pool; then the encoder MLP.
__global__ __launch_bounds__(256) void k_encode(
    const float* __restrict__ xc, const float* __restrict__ yc,
    const float* __restrict__ w1, const float* __restrict__ b1,
    const float* __restrict__ b2,
    const float* __restrict__ We1, const float* __restrict__ be1,
    const float* __restrict__ We2, const float* __restrict__ be2,
    const unsigned short* __restrict__ wf2, float* __restrict__ racc)
{
  extern __shared__ char smem[];
  unsigned short* hbuf = (unsigned short*)smem;                 // 52*52*32 bf16 = 173056 B
  float* yg     = (float*)(smem + 173056);                      // 2500 f32
  float* pooled = (float*)(smem + 183056);                      // 32 f32
  float* hm     = (float*)(smem + 183184);                      // 128 f32

  const int s = blockIdx.x;
  const int tid = threadIdx.x;
  const float xv = xc[s];

  for (int i = tid; i < (52 * 52 * 32) / 2; i += 256) ((unsigned int*)hbuf)[i] = 0u;
  for (int i = tid; i < YD; i += 256) yg[i] = yc[(long)s * YD + i];
  if (tid < CC) pooled[tid] = 0.f;
  __syncthreads();

  // conv1 + relu
  for (int i = tid; i < CC * YD; i += 256) {
    int c = i / YD;
    int p = i - c * YD;
    int py = p / GRIDW, px = p - py * GRIDW;
    float acc = b1[c];
    for (int dy = 0; dy < 3; ++dy) {
      int iy = py + dy - 1;
      if ((unsigned)iy >= GRIDW) continue;
      for (int dx = 0; dx < 3; ++dx) {
        int ix = px + dx - 1;
        if ((unsigned)ix >= GRIDW) continue;
        acc += w1[((c * 2 + 0) * 3 + dy) * 3 + dx] * yg[iy * GRIDW + ix]
             + w1[((c * 2 + 1) * 3 + dy) * 3 + dx] * xv;
      }
    }
    hbuf[((py + 1) * 52 + (px + 1)) * CC + c] = f2bf(fmaxf(acc, 0.f));
  }
  __syncthreads();

  // conv2 implicit GEMM: M=2500 px (157 tiles), N=32 couts (2 tiles), K=9 taps x 32 cin
  const int lane = tid & 31;
  const int wave = tid >> 5;
  const int m16  = lane & 15;
  const int g    = lane >> 4;
  const float bias0 = b2[m16];
  const float bias1 = b2[16 + m16];

  // Hoist all 18 B-fragments (9 taps x 2 N-tiles) into registers (loop-invariant).
  v16bf bw[9][2];
#pragma unroll
  for (int tap = 0; tap < 9; ++tap) {
#pragma unroll
    for (int nt = 0; nt < 2; ++nt) {
      const unsigned short* wb = wf2 + ((tap * 2 + nt) * 32 + lane) * 16;
      bw[tap][nt] = mk_frag(*(const uint4*)(wb), *(const uint4*)(wb + 8));
    }
  }

  for (int tile = wave; tile < NTILES; tile += 8) {
    v8f acc0 = {0.f, 0.f, 0.f, 0.f, 0.f, 0.f, 0.f, 0.f};
    v8f acc1 = {0.f, 0.f, 0.f, 0.f, 0.f, 0.f, 0.f, 0.f};
    const int pix = tile * 16 + m16;
    const unsigned vm = (pix < YD) ? 0xFFFFFFFFu : 0u;  // lane-mask, no branches
    const int pp = (pix < YD) ? pix : 0;                // safe LDS address for dead lanes
    const int py = pp / GRIDW, px = pp - py * GRIDW;
    const unsigned short* pbase = hbuf + (py * 52 + px) * CC;

#pragma unroll
    for (int tap = 0; tap < 9; ++tap) {
      const int dy = tap / 3, dx = tap - dy * 3;
      const unsigned short* base = pbase + (dy * 52 + dx) * CC;
      uint4 q0 = *(const uint4*)(base + g * 16);
      uint4 q1 = *(const uint4*)(base + g * 16 + 8);
      q0.x &= vm; q0.y &= vm; q0.z &= vm; q0.w &= vm;
      q1.x &= vm; q1.y &= vm; q1.z &= vm; q1.w &= vm;
      v16bf av = mk_frag(q0, q1);
      acc0 = __builtin_amdgcn_wmma_f32_16x16x32_bf16(false, av, false, bw[tap][0], (short)0, acc0, false, false);
      acc1 = __builtin_amdgcn_wmma_f32_16x16x32_bf16(false, av, false, bw[tap][1], (short)0, acc1, false, false);
    }
    // bias + relu + pool (conv2 output never leaves registers)
    float s0 = 0.f, s1 = 0.f;
#pragma unroll
    for (int r = 0; r < 8; ++r) {
      int p2 = tile * 16 + r + 8 * g;
      if (p2 < YD) {
        s0 += fmaxf(acc0[r] + bias0, 0.f);
        s1 += fmaxf(acc1[r] + bias1, 0.f);
      }
    }
    atomicAdd(&pooled[m16], s0);
    atomicAdd(&pooled[16 + m16], s1);
  }
  __syncthreads();

  // encoder MLP: feat[33] = {pooled/2500, x}; h = relu(feat@We1+be1); r_i = h@We2+be2
  if (tid < HD) {
    float acc = be1[tid];
    for (int i = 0; i < CC; ++i) acc += (pooled[i] * (1.f / YD)) * We1[i * HD + tid];
    acc += xv * We1[CC * HD + tid];
    hm[tid] = fmaxf(acc, 0.f);
  }
  __syncthreads();
  if (tid < RD) {
    float acc = be2[tid];
    for (int i = 0; i < HD; ++i) acc += hm[i] * We2[i * RD + tid];
    atomicAdd(&racc[(s >> 6) * RD + tid], acc * (1.f / NCV));
  }
}

// Latent head: hr = relu(r@Wh+bh); mu/sigma; z = mu + sigma*eps. Tiny -> VALU.
__global__ __launch_bounds__(256) void k_latent(
    const float* __restrict__ racc, const float* __restrict__ eps,
    const float* __restrict__ Wh,  const float* __restrict__ bh,
    const float* __restrict__ Wmu, const float* __restrict__ bmu,
    const float* __restrict__ Wsig, const float* __restrict__ bsig,
    float* __restrict__ out, float* __restrict__ zws)
{
  __shared__ float hr[BB * RD];
  const int tid = threadIdx.x;
  for (int idx = tid; idx < BB * RD; idx += 256) {
    int b = idx >> 7, t = idx & 127;
    float acc = bh[t];
    for (int i = 0; i < RD; ++i) acc += racc[b * RD + i] * Wh[i * RD + t];
    hr[idx] = fmaxf(acc, 0.f);
  }
  __syncthreads();
  const long OUTY = (long)BB * NTGT * YD;  // 5,120,000
  for (int idx = tid; idx < BB * ZD; idx += 256) {
    int b = idx >> 6, t = idx & 63;
    float mu = bmu[t], sl = bsig[t];
    for (int i = 0; i < RD; ++i) {
      float h = hr[b * RD + i];
      mu += h * Wmu[i * ZD + t];
      sl += h * Wsig[i * ZD + t];
    }
    float sg = 0.1f + 0.9f / (1.f + expf(-sl));
    out[2 * OUTY + idx]           = mu;
    out[2 * OUTY + BB * ZD + idx] = sg;
    zws[idx] = mu + sg * eps[idx];
  }
}

// Decoder hidden layers (tiny GEMMs -> VALU), store h2 as bf16 for the WMMA tail.
__global__ __launch_bounds__(128) void k_dec_hidden(
    const float* __restrict__ xt, const float* __restrict__ zws,
    const float* __restrict__ Wd1, const float* __restrict__ bd1,
    const float* __restrict__ Wd2, const float* __restrict__ bd2,
    unsigned short* __restrict__ h2)
{
  __shared__ float xz[1 + ZD];
  __shared__ float h1[HD];
  const int row = blockIdx.x;       // b*128 + m
  const int t = threadIdx.x;
  const int b = row >> 7;
  if (t == 0) xz[0] = xt[row];
  if (t < ZD) xz[1 + t] = zws[b * ZD + t];
  __syncthreads();
  float acc = bd1[t];
  for (int i = 0; i < 1 + ZD; ++i) acc += xz[i] * Wd1[i * HD + t];
  h1[t] = fmaxf(acc, 0.f);
  __syncthreads();
  float acc2 = bd2[t];
  for (int i = 0; i < HD; ++i) acc2 += h1[i] * Wd2[i * HD + t];
  h2[(long)row * HD + t] = f2bf(fmaxf(acc2, 0.f));
}

// Decoder output GEMM: [2048,128] @ [128,2500] for mu and sigma, shared A frags.
__global__ __launch_bounds__(256) void k_dec_out(
    const unsigned short* __restrict__ h2,
    const unsigned short* __restrict__ fmu, const unsigned short* __restrict__ fsig,
    const float* __restrict__ bdmu, const float* __restrict__ bdsig,
    float* __restrict__ out)
{
  const int lane  = threadIdx.x & 31;
  const int wave  = threadIdx.x >> 5;
  const int mtile = blockIdx.x;
  const int ntile = blockIdx.y * 8 + wave;
  if (ntile >= NTILES) return;               // wave-uniform
  const int m16 = lane & 15;
  const int g   = lane >> 4;

  // A fragments: 4 K-blocks from the bf16 h2 rows (shared by mu & sigma GEMMs)
  v16bf a[4];
  const unsigned short* rowp = h2 + (long)(mtile * 16 + m16) * HD;
#pragma unroll
  for (int kb = 0; kb < 4; ++kb) {
    a[kb] = mk_frag(*(const uint4*)(rowp + kb * 32 + g * 16),
                    *(const uint4*)(rowp + kb * 32 + g * 16 + 8));
  }

  v8f am = {0.f, 0.f, 0.f, 0.f, 0.f, 0.f, 0.f, 0.f};
  v8f as = {0.f, 0.f, 0.f, 0.f, 0.f, 0.f, 0.f, 0.f};
#pragma unroll
  for (int kb = 0; kb < 4; ++kb) {
    const unsigned short* pm = fmu  + ((ntile * 4 + kb) * 32 + lane) * 16;
    const unsigned short* ps = fsig + ((ntile * 4 + kb) * 32 + lane) * 16;
    __builtin_prefetch(pm + 4 * 32 * 16, 0, 0);   // next ntile's fragments
    v16bf bm = mk_frag(*(const uint4*)pm, *(const uint4*)(pm + 8));
    v16bf bs = mk_frag(*(const uint4*)ps, *(const uint4*)(ps + 8));
    am = __builtin_amdgcn_wmma_f32_16x16x32_bf16(false, a[kb], false, bm, (short)0, am, false, false);
    as = __builtin_amdgcn_wmma_f32_16x16x32_bf16(false, a[kb], false, bs, (short)0, as, false, false);
  }
  const long OUTY = (long)BB * NTGT * YD;
  const int col = ntile * 16 + m16;
  if (col < YD) {
    const float bmu_ = bdmu[col], bsg_ = bdsig[col];
#pragma unroll
    for (int r = 0; r < 8; ++r) {
      int row = mtile * 16 + r + 8 * g;
      float mu = am[r] + bmu_;
      float x  = as[r] + bsg_;
      float sp = (x > 20.f) ? x : log1pf(expf(x));
      out[(long)row * YD + col]        = mu;
      out[OUTY + (long)row * YD + col] = 0.1f + 0.9f * sp;
    }
  }
}

extern "C" void kernel_launch(void* const* d_in, const int* in_sizes, int n_in,
                              void* d_out, int out_size, void* d_ws, size_t ws_size,
                              hipStream_t stream) {
  (void)in_sizes; (void)n_in; (void)out_size; (void)ws_size;
  const float* xc   = (const float*)d_in[0];
  const float* yc   = (const float*)d_in[1];
  const float* xt   = (const float*)d_in[2];
  const float* eps  = (const float*)d_in[3];
  const float* w1   = (const float*)d_in[4];
  const float* b1   = (const float*)d_in[5];
  const float* w2   = (const float*)d_in[6];
  const float* b2   = (const float*)d_in[7];
  const float* We1  = (const float*)d_in[8];
  const float* be1  = (const float*)d_in[9];
  const float* We2  = (const float*)d_in[10];
  const float* be2  = (const float*)d_in[11];
  const float* Wh   = (const float*)d_in[12];
  const float* bh   = (const float*)d_in[13];
  const float* Wmu  = (const float*)d_in[14];
  const float* bmu  = (const float*)d_in[15];
  const float* Wsig = (const float*)d_in[16];
  const float* bsig = (const float*)d_in[17];
  const float* Wd1  = (const float*)d_in[18];
  const float* bd1  = (const float*)d_in[19];
  const float* Wd2  = (const float*)d_in[20];
  const float* bd2  = (const float*)d_in[21];
  const float* Wdmu = (const float*)d_in[22];
  const float* bdmu = (const float*)d_in[23];
  const float* Wdsg = (const float*)d_in[24];
  const float* bdsg = (const float*)d_in[25];

  char* ws = (char*)d_ws;
  float* racc          = (float*)(ws + 0);
  float* zws           = (float*)(ws + 8192);
  unsigned short* h2   = (unsigned short*)(ws + 12288);
  unsigned short* wf2  = (unsigned short*)(ws + 536576);
  unsigned short* fmu  = (unsigned short*)(ws + 555008);
  unsigned short* fsig = (unsigned short*)(ws + 1198080);

  k_init<<<8, 256, 0, stream>>>(racc);
  k_prep_w2<<<36, 256, 0, stream>>>(w2, wf2);
  k_prep_wout<<<(2 * NTILES * 4 * 32 * 16 + 255) / 256, 256, 0, stream>>>(Wdmu, Wdsg, fmu, fsig);
  k_encode<<<BB * NCV, 256, 184320, stream>>>(xc, yc, w1, b1, b2, We1, be1, We2, be2, wf2, racc);
  k_latent<<<1, 256, 0, stream>>>(racc, eps, Wh, bh, Wmu, bmu, Wsig, bsig, (float*)d_out, zws);
  k_dec_hidden<<<BB * NTGT, 128, 0, stream>>>(xt, zws, Wd1, bd1, Wd2, bd2, h2);
  k_dec_out<<<dim3(128, 20), 256, 0, stream>>>(h2, fmu, fsig, bdmu, bdsg, (float*)d_out);
}
